// Transformer4d_75402445848965
// MI455X (gfx1250) — compile-verified, matrix operand order checked by reading, not run on hardware
//
#include <hip/hip_runtime.h>
#include <hip/hip_bf16.h>
#include <math.h>

// ---------------------------------------------------------------------------
// Transformer block for MI455X (gfx1250, wave32, WMMA + Tensor Data Mover).
// All GEMMs via v_wmma_f32_16x16x32_f16 (f16 in, f32 accum); GEMM tiles are
// staged into LDS by the TDM (tensor_load_to_lds, TENSORcnt), with the TDM's
// pad feature reproducing the +16B/row LDS padding.
// Workspace requirement: ~230 MB (harness d_ws).
// ---------------------------------------------------------------------------

typedef _Float16 f16_t;
typedef __attribute__((ext_vector_type(8)))  _Float16     v8h;
typedef __attribute__((ext_vector_type(16))) _Float16     v16h;
typedef __attribute__((ext_vector_type(8)))  float        v8f;
typedef __attribute__((ext_vector_type(4)))  unsigned int u32x4;
typedef __attribute__((ext_vector_type(4)))  int          i32x4;
typedef __attribute__((ext_vector_type(8)))  int          i32x8;

#define TOKENS 16384   // B*F*N = 2*64*128
#define DIM    768
#define HIDDEN 3072
#define HEADS  8
#define DEPTH  96
#define SEQ    128

// ---------------------------------------------------------------------------
// WMMA fragment helper. Per ISA 7.12.2 (16-bit A/B operand 16x32): lane L
// (half-group hg = L>>4) holds K = [8hg..8hg+7] and [16+8hg..16+8hg+7] of its
// row/column, so a fragment is two contiguous 16B LDS loads.
// ---------------------------------------------------------------------------
__device__ __forceinline__ v16h frag16(const f16_t* rowp, int hg) {
  v8h lo = *(const v8h*)(rowp + 8 * hg);
  v8h hi = *(const v8h*)(rowp + 16 + 8 * hg);
  v16h r;
#pragma unroll
  for (int i = 0; i < 8; ++i) { r[i] = lo[i]; r[i + 8] = hi[i]; }
  return r;
}

__device__ __forceinline__ v8f wmma16(v16h a, v16h b, v8f c) {
  return __builtin_amdgcn_wmma_f32_16x16x32_f16(false, a, false, b,
                                                (short)0, c, false, false);
}

// ---------------------------------------------------------------------------
// TDM: issue a 2D tensor tile load into LDS (wave-level DMA, TENSORcnt).
// D# per ISA 08_async_tensor §8.3/8.4. data_size = 2B (f16).
// tile: tile_d1 rows x tile_d0 elements, global row stride = stride_d0 elems.
// Optional LDS padding: pad_amount_code+1 DWORDs every 2^(pad_interval_code+1)
// DWORDs stored.
// ---------------------------------------------------------------------------
__device__ __forceinline__ void tdm_load_tile2d(
    const f16_t* gsrc, unsigned lds_off, unsigned tensor_d0, unsigned tensor_d1,
    unsigned tile_d0, unsigned tile_d1, unsigned stride_d0, bool pad_en,
    unsigned pad_interval_code, unsigned pad_amount_code) {
  unsigned long long ga = (unsigned long long)gsrc;
  u32x4 g0;
  g0[0] = 1u;                                      // count=1 (user descriptor)
  g0[1] = lds_off;                                 // lds_addr (bytes)
  g0[2] = (unsigned)(ga & 0xFFFFFFFFu);            // global_addr[31:0]
  g0[3] = (unsigned)((ga >> 32) & 0x1FFFFFFu)      // global_addr[56:32]
          | (2u << 30);                            // type = 2 ("image")
  i32x8 g1;
  g1[0] = (int)((1u << 16)                               // data_size = 2B
                | (pad_en ? (1u << 20) : 0u)             // pad_enable
                | (pad_interval_code << 22)
                | (pad_amount_code << 25));
  g1[1] = (int)((tensor_d0 & 0xFFFFu) << 16);            // tensor_dim0 lo
  g1[2] = (int)(((tensor_d0 >> 16) & 0xFFFFu)            // tensor_dim0 hi
                | ((tensor_d1 & 0xFFFFu) << 16));        // tensor_dim1 lo
  g1[3] = (int)(((tensor_d1 >> 16) & 0xFFFFu)            // tensor_dim1 hi
                | ((tile_d0 & 0xFFFFu) << 16));          // tile_dim0
  g1[4] = (int)(tile_d1 & 0xFFFFu);                      // tile_dim1 (dim2=0)
  g1[5] = (int)stride_d0;                                // dim0 stride lo
  g1[6] = 0;                                             // stride hi, dim1 stride
  g1[7] = 0;
  i32x4 z4 = {0, 0, 0, 0};
#if __has_include(<hip/amd_detail/amd_gfx1250_TDM.h>)
  i32x8 z8 = {0, 0, 0, 0, 0, 0, 0, 0};
  __builtin_amdgcn_tensor_load_to_lds(g0, g1, z4, z4, z8, 0);   // clang-23 form
#else
  __builtin_amdgcn_tensor_load_to_lds(g0, g1, z4, z4, 0);       // ROCm 7.2 form
#endif
}

// ---------------------------------------------------------------------------
// f32 -> f16 cast with transpose: dst[n*K + k] = (f16)src[k*N + n].
// Weights are pre-transposed so GEMM B tiles are row-major [n][k] (TDM- and
// fragment-friendly: no transpose anywhere in the hot path).
// ---------------------------------------------------------------------------
__global__ __launch_bounds__(256)
void cast_transpose_kernel(const float* __restrict__ src, f16_t* __restrict__ dst,
                           int K, int N) {
  int i = blockIdx.x * 256 + threadIdx.x;
  if (i < K * N) {
    int kk = i / N, nn = i % N;
    dst[(size_t)nn * K + kk] = (f16_t)src[i];
  }
}

// ---------------------------------------------------------------------------
// Block-wide row statistics (768 elems, 256 threads = 8 waves)
// ---------------------------------------------------------------------------
__device__ __forceinline__ void row_stats(float s, float s2,
                                          float* tmp /*[16] shared*/,
                                          float* stats /*[2] shared*/) {
  const int tid = threadIdx.x;
#pragma unroll
  for (int off = 16; off > 0; off >>= 1) {
    s  += __shfl_xor(s, off, 32);
    s2 += __shfl_xor(s2, off, 32);
  }
  if ((tid & 31) == 0) { tmp[tid >> 5] = s; tmp[8 + (tid >> 5)] = s2; }
  __syncthreads();
  if (tid == 0) {
    float ts = 0.f, ts2 = 0.f;
#pragma unroll
    for (int i = 0; i < 8; ++i) { ts += tmp[i]; ts2 += tmp[8 + i]; }
    float mu  = ts * (1.0f / (float)DIM);
    float var = ts2 * (1.0f / (float)DIM) - mu * mu;
    stats[0] = mu;
    stats[1] = rsqrtf(var + 1e-5f);
  }
  __syncthreads();
}

// ---------------------------------------------------------------------------
// LN1: normalize q,k,v rows (shared gamma/beta); emit f16 GEMM operands and
// qn in f32 (residual). One token per block.
// ---------------------------------------------------------------------------
__global__ __launch_bounds__(256)
void ln1_kernel(const float* __restrict__ q, const float* __restrict__ k,
                const float* __restrict__ v, const float* __restrict__ g,
                const float* __restrict__ b, float* __restrict__ qnF,
                f16_t* __restrict__ qn16, f16_t* __restrict__ kn16,
                f16_t* __restrict__ vn16) {
  __shared__ float tmp[16];
  __shared__ float stats[2];
  const int tid = threadIdx.x;
  const size_t base = (size_t)blockIdx.x * DIM;
#pragma unroll
  for (int m = 0; m < 3; ++m) {
    const float* src = (m == 0 ? q : (m == 1 ? k : v)) + base;
    float x0 = src[tid], x1 = src[tid + 256], x2 = src[tid + 512];
    row_stats(x0 + x1 + x2, x0 * x0 + x1 * x1 + x2 * x2, tmp, stats);
    const float mu = stats[0], rs = stats[1];
    float y0 = (x0 - mu) * rs * g[tid]       + b[tid];
    float y1 = (x1 - mu) * rs * g[tid + 256] + b[tid + 256];
    float y2 = (x2 - mu) * rs * g[tid + 512] + b[tid + 512];
    f16_t* dst = (m == 0 ? qn16 : (m == 1 ? kn16 : vn16)) + base;
    dst[tid] = (f16_t)y0; dst[tid + 256] = (f16_t)y1; dst[tid + 512] = (f16_t)y2;
    if (m == 0) {
      qnF[base + tid] = y0; qnF[base + tid + 256] = y1; qnF[base + tid + 512] = y2;
    }
    __syncthreads();
  }
}

// ---------------------------------------------------------------------------
// LN2: y(f32) -> hn(f16). One row per block.
// ---------------------------------------------------------------------------
__global__ __launch_bounds__(256)
void ln2_kernel(const float* __restrict__ y, const float* __restrict__ g,
                const float* __restrict__ b, f16_t* __restrict__ out) {
  __shared__ float tmp[16];
  __shared__ float stats[2];
  const int tid = threadIdx.x;
  const size_t base = (size_t)blockIdx.x * DIM;
  float x0 = y[base + tid], x1 = y[base + tid + 256], x2 = y[base + tid + 512];
  row_stats(x0 + x1 + x2, x0 * x0 + x1 * x1 + x2 * x2, tmp, stats);
  const float mu = stats[0], rs = stats[1];
  out[base + tid]       = (f16_t)((x0 - mu) * rs * g[tid]       + b[tid]);
  out[base + tid + 256] = (f16_t)((x1 - mu) * rs * g[tid + 256] + b[tid + 256]);
  out[base + tid + 512] = (f16_t)((x2 - mu) * rs * g[tid + 512] + b[tid + 512]);
}

// ---------------------------------------------------------------------------
// TDM-staged WMMA GEMM:  C[MxN] = A[MxK] * BT[NxK]^T (+bias)(gelu)(+res)
//
// Block tile 128(M) x 128(N), K step 32, double-buffered LDS. Wave 0 issues
// two tensor_load_to_lds per K step (A tile 128x32 and BT tile 128x32, both
// row-major, TDM pads each 64B row with 16B -> padded stride GK=40 f16);
// prefetch of tile i+1 overlaps WMMA work on tile i. 8 waves in a 4x2 grid,
// each wave owns a 32x64 strip = 8 accumulator tiles -> 8 WMMA per K step.
// Requires M%128==0, N%128==0, K%32==0 (true for all six calls).
// ---------------------------------------------------------------------------
#define GK 40  // padded f16 K-stride in LDS (32 data + 8 pad)

template <bool HAS_BIAS, bool HAS_RES, bool DO_GELU, bool OUT_F32>
__global__ __launch_bounds__(256)
void wmma_gemm_kernel(const f16_t* __restrict__ A, const f16_t* __restrict__ BT,
                      const float* __restrict__ bias, const float* __restrict__ res,
                      float* __restrict__ outF, f16_t* __restrict__ outH,
                      int M, int N, int K) {
  __shared__ __align__(16) f16_t As[2][128 * GK];   // [buf][row][k]
  __shared__ __align__(16) f16_t Bs[2][128 * GK];   // [buf][col][k]

  const int tid  = threadIdx.x;
  const int lane = tid & 31;
  const int la   = lane & 15;
  const int hg   = lane >> 4;
  const int wave = tid >> 5;
  const int wm   = wave >> 1;   // 0..3 -> rows wm*32
  const int wn   = wave & 1;    // 0..1 -> cols wn*64
  const int m0   = blockIdx.y * 128;
  const int n0   = blockIdx.x * 128;

  v8f acc[2][4] = {};

  const int nk = K / 32;
  if (wave == 0) {  // wave-uniform branch: exactly one wave drives the TDM
    tdm_load_tile2d(A  + (size_t)m0 * K, (unsigned)(size_t)&As[0][0],
                    (unsigned)K, 1u << 20, 32, 128, (unsigned)K, true, 3, 3);
    tdm_load_tile2d(BT + (size_t)n0 * K, (unsigned)(size_t)&Bs[0][0],
                    (unsigned)K, 1u << 20, 32, 128, (unsigned)K, true, 3, 3);
  }

  for (int it = 0; it < nk; ++it) {
    const int cur = it & 1;
    if (wave == 0) __builtin_amdgcn_s_wait_tensorcnt(0);  // tiles(it) landed
    __syncthreads();
    if (wave == 0 && it + 1 < nk) {  // prefetch tiles(it+1) into other buffer
      const int kb = (it + 1) * 32;
      tdm_load_tile2d(A  + (size_t)m0 * K + kb, (unsigned)(size_t)&As[cur ^ 1][0],
                      (unsigned)K, 1u << 20, 32, 128, (unsigned)K, true, 3, 3);
      tdm_load_tile2d(BT + (size_t)n0 * K + kb, (unsigned)(size_t)&Bs[cur ^ 1][0],
                      (unsigned)K, 1u << 20, 32, 128, (unsigned)K, true, 3, 3);
    }

    v16h a0 = frag16(&As[cur][(wm * 32 + la) * GK], hg);
    v16h a1 = frag16(&As[cur][(wm * 32 + 16 + la) * GK], hg);
#pragma unroll
    for (int t = 0; t < 4; ++t) {
      v16h bfr = frag16(&Bs[cur][(wn * 64 + t * 16 + la) * GK], hg);
      acc[0][t] = wmma16(a0, bfr, acc[0][t]);
      acc[1][t] = wmma16(a1, bfr, acc[1][t]);
    }
    __syncthreads();  // all waves done with buf `cur` before it is refilled
  }

  // epilogue: D layout — VGPR v, lane L: row = v + 8*(L>>4), col = L&15
#pragma unroll
  for (int s = 0; s < 2; ++s) {
#pragma unroll
    for (int t = 0; t < 4; ++t) {
#pragma unroll
      for (int vv = 0; vv < 8; ++vv) {
        const int row = m0 + wm * 32 + s * 16 + vv + 8 * hg;
        const int col = n0 + wn * 64 + t * 16 + la;
        float x = acc[s][t][vv];
        if (HAS_BIAS) x += bias[col];
        if (DO_GELU) {
          const float c0 = 0.7978845608028654f;
          x = 0.5f * x * (1.0f + tanhf(c0 * (x + 0.044715f * x * x * x)));
        }
        if (HAS_RES) x += res[(size_t)row * N + col];
        if (OUT_F32) outF[(size_t)row * N + col] = x;
        else         outH[(size_t)row * N + col] = (f16_t)x;
      }
    }
  }
}

// ---------------------------------------------------------------------------
// Fused attention per (b,f,head): S = Q K^T * scale, softmax, O = P V.
// Q/K/V/O stored [token, h*96+d] (row stride 768). One block per head
// instance, 256 threads = 8 waves; wave w owns a 16-row strip of S/P/O.
// LDS: Qs/Ks row-major [128][96+8], Vt transposed [96][128+8], Ps [128][128+8]
// overlapping the Qs/Ks region (dead after S compute).
// ---------------------------------------------------------------------------
#define QK_STRIDE 104   // 96 + 8 pad
#define VT_STRIDE 136   // 128 + 8 pad
#define PS_STRIDE 136

__global__ __launch_bounds__(256)
void attn_kernel(const f16_t* __restrict__ Qh, const f16_t* __restrict__ Kh,
                 const f16_t* __restrict__ Vh, float* __restrict__ att_out,
                 f16_t* __restrict__ Oout) {
  __shared__ __align__(16) f16_t smem[39680];  // 79,360 B
  f16_t* Qs = smem;                     // 128*104 = 13312
  f16_t* Ks = smem + 13312;             // 128*104
  f16_t* Vt = smem + 26624;             // 96*136 = 13056
  f16_t* Ps = smem;                     // 128*136 = 17408 (overlaps Qs/Ks)

  const int tid  = threadIdx.x;
  const int lane = tid & 31;
  const int la   = lane & 15;
  const int hg   = lane >> 4;
  const int wave = tid >> 5;

  const int bfh  = blockIdx.x;       // (b*F+f)*8 + head
  const int bf   = bfh >> 3;
  const int head = bfh & 7;

  // ---- stage Q, K, V^T into LDS: thread t loads row t/2, cols (t&1)*48..+47
  {
    const int row   = tid >> 1;
    const int cbase = (tid & 1) * 48;
    const size_t gbase = ((size_t)bf * SEQ + row) * DIM + head * DEPTH + cbase;
#pragma unroll
    for (int j = 0; j < 6; ++j) {
      v8h xq = *(const v8h*)(Qh + gbase + j * 8);
      v8h xk = *(const v8h*)(Kh + gbase + j * 8);
      v8h xv = *(const v8h*)(Vh + gbase + j * 8);
      *(v8h*)(&Qs[row * QK_STRIDE + cbase + j * 8]) = xq;
      *(v8h*)(&Ks[row * QK_STRIDE + cbase + j * 8]) = xk;
#pragma unroll
      for (int i = 0; i < 8; ++i)
        Vt[(cbase + j * 8 + i) * VT_STRIDE + row] = xv[i];
    }
  }
  __syncthreads();

  // ---- S = Q K^T : wave strip rows 16w, 8 col tiles, K loop over DEPTH=96
  v8f s[8] = {};
  for (int d0 = 0; d0 < DEPTH; d0 += 32) {
    v16h af = frag16(&Qs[(wave * 16 + la) * QK_STRIDE + d0], hg);
#pragma unroll
    for (int t = 0; t < 8; ++t) {
      v16h bfr = frag16(&Ks[(t * 16 + la) * QK_STRIDE + d0], hg);
      s[t] = wmma16(af, bfr, s[t]);
    }
  }

  // ---- softmax over each row (a row lives in one 16-lane half-group)
  const float scale = 0.10206207261596575f;  // 1/sqrt(96)
  float inv_sum[8];
#pragma unroll
  for (int vv = 0; vv < 8; ++vv) {
    float m = -3.0e38f;
#pragma unroll
    for (int t = 0; t < 8; ++t) { s[t][vv] *= scale; m = fmaxf(m, s[t][vv]); }
#pragma unroll
    for (int off = 1; off < 16; off <<= 1) m = fmaxf(m, __shfl_xor(m, off, 32));
    float sum = 0.f;
#pragma unroll
    for (int t = 0; t < 8; ++t) {
      float p = __expf(s[t][vv] - m);
      s[t][vv] = p;
      sum += p;
    }
#pragma unroll
    for (int off = 1; off < 16; off <<= 1) sum += __shfl_xor(sum, off, 32);
    inv_sum[vv] = 1.0f / sum;
  }

  __syncthreads();  // all waves done reading Qs/Ks before Ps overwrites them

  // ---- write att_weight (f32, output #2) + P (f16) into LDS for re-layout
  const size_t abase = (size_t)bfh * (SEQ * SEQ);
#pragma unroll
  for (int t = 0; t < 8; ++t) {
#pragma unroll
    for (int vv = 0; vv < 8; ++vv) {
      const int r = wave * 16 + vv + 8 * hg;
      const int c = t * 16 + la;
      const float p = s[t][vv] * inv_sum[vv];
      att_out[abase + (size_t)r * SEQ + c] = p;
      Ps[r * PS_STRIDE + c] = (f16_t)p;
    }
  }
  __syncthreads();

  // ---- O = P V : 6 col tiles of DEPTH, K loop over SEQ=128
  v8f o[6] = {};
  for (int j0 = 0; j0 < SEQ; j0 += 32) {
    v16h af = frag16(&Ps[(wave * 16 + la) * PS_STRIDE + j0], hg);
#pragma unroll
    for (int td = 0; td < 6; ++td) {
      v16h bfr = frag16(&Vt[(td * 16 + la) * VT_STRIDE + j0], hg);
      o[td] = wmma16(af, bfr, o[td]);
    }
  }

  // ---- store O in [token, h*96+d] f16
#pragma unroll
  for (int td = 0; td < 6; ++td) {
#pragma unroll
    for (int vv = 0; vv < 8; ++vv) {
      const int r = wave * 16 + vv + 8 * hg;
      const int d = td * 16 + la;
      Oout[((size_t)bf * SEQ + r) * DIM + head * DEPTH + d] = (f16_t)o[td][vv];
    }
  }
}

// ---------------------------------------------------------------------------
// Host-side orchestration
// ---------------------------------------------------------------------------
extern "C" void kernel_launch(void* const* d_in, const int* in_sizes, int n_in,
                              void* d_out, int out_size, void* d_ws, size_t ws_size,
                              hipStream_t stream) {
  const float* q     = (const float*)d_in[0];
  const float* k     = (const float*)d_in[1];
  const float* v     = (const float*)d_in[2];
  const float* ln1_g = (const float*)d_in[3];
  const float* ln1_b = (const float*)d_in[4];
  const float* wq    = (const float*)d_in[5];
  const float* wk    = (const float*)d_in[6];
  const float* wv    = (const float*)d_in[7];
  const float* wo    = (const float*)d_in[8];
  const float* wo_b  = (const float*)d_in[9];
  const float* ln2_g = (const float*)d_in[10];
  const float* ln2_b = (const float*)d_in[11];
  const float* w1    = (const float*)d_in[12];
  const float* b1    = (const float*)d_in[13];
  const float* w2    = (const float*)d_in[14];
  const float* b2    = (const float*)d_in[15];

  char* ws = (char*)d_ws;
  const size_t TD   = (size_t)TOKENS * DIM;          // 12,582,912 elems
  const size_t SZ32 = TD * 4;                        // 50,331,648 B
  const size_t SZ16 = TD * 2;                        // 25,165,824 B

  // ---- workspace layout (total 240,648,192 B) with liveness-based reuse
  float* qnF  = (float*)(ws);                        // [0, SZ32)
  f16_t* qn16 = (f16_t*)(ws + SZ32);                 // abc16 region
  f16_t* kn16 = qn16 + TD;
  f16_t* vn16 = kn16 + TD;
  f16_t* Qh   = (f16_t*)(ws + SZ32 + 3 * SZ16);      // qkv-head region
  f16_t* Kh   = Qh + TD;
  f16_t* Vh   = Kh + TD;
  f16_t* O16  = (f16_t*)(ws + SZ32 + 6 * SZ16);      // attention output
  f16_t* W16  = (f16_t*)(ws + SZ32 + 7 * SZ16);      // f16 transposed weights
  f16_t* wqT  = W16;                                 // [N][K] layouts
  f16_t* wkT  = wqT + (size_t)DIM * DIM;
  f16_t* wvT  = wkT + (size_t)DIM * DIM;
  f16_t* woT  = wvT + (size_t)DIM * DIM;
  f16_t* w1T  = woT + (size_t)DIM * DIM;             // [HIDDEN][DIM]
  f16_t* w2T  = w1T + (size_t)DIM * HIDDEN;          // [DIM][HIDDEN]
  // reuse (dead regions):
  float* yF   = (float*)qn16;            // overlays qn16+kn16 (dead post-QKV)
  f16_t* hn16 = (f16_t*)((char*)qn16 + SZ32);   // overlays vn16
  f16_t* h16  = Qh;                      // overlays Qh..O16 (dead post-attn/wo)

  float* outF    = (float*)d_out;        // out2 + y   : TOKENS*DIM f32
  float* att_out = outF + TD;            // att_weight : 1024*128*128 f32

  // ---- 1. weight cast + transpose (B operands become row-major [n][k])
  {
    const int nd = DIM * DIM, nh = DIM * HIDDEN;
    cast_transpose_kernel<<<(nd + 255) / 256, 256, 0, stream>>>(wq, wqT, DIM, DIM);
    cast_transpose_kernel<<<(nd + 255) / 256, 256, 0, stream>>>(wk, wkT, DIM, DIM);
    cast_transpose_kernel<<<(nd + 255) / 256, 256, 0, stream>>>(wv, wvT, DIM, DIM);
    cast_transpose_kernel<<<(nd + 255) / 256, 256, 0, stream>>>(wo, woT, DIM, DIM);
    cast_transpose_kernel<<<(nh + 255) / 256, 256, 0, stream>>>(w1, w1T, DIM, HIDDEN);
    cast_transpose_kernel<<<(nh + 255) / 256, 256, 0, stream>>>(w2, w2T, HIDDEN, DIM);
  }

  // ---- 2. LN1 (q,k,v) -> f16 operands + qn f32 residual
  ln1_kernel<<<TOKENS, 256, 0, stream>>>(q, k, v, ln1_g, ln1_b,
                                         qnF, qn16, kn16, vn16);

  // ---- 3. QKV projections (f16 out, head layout = natural [token, h*96+d])
  dim3 gproj(DIM / 128, TOKENS / 128);
  wmma_gemm_kernel<false, false, false, false><<<gproj, 256, 0, stream>>>(
      qn16, wqT, nullptr, nullptr, nullptr, Qh, TOKENS, DIM, DIM);
  wmma_gemm_kernel<false, false, false, false><<<gproj, 256, 0, stream>>>(
      kn16, wkT, nullptr, nullptr, nullptr, Kh, TOKENS, DIM, DIM);
  wmma_gemm_kernel<false, false, false, false><<<gproj, 256, 0, stream>>>(
      vn16, wvT, nullptr, nullptr, nullptr, Vh, TOKENS, DIM, DIM);

  // ---- 4. fused attention (writes att_weight f32 + O f16)
  attn_kernel<<<2 * 64 * HEADS, 256, 0, stream>>>(Qh, Kh, Vh, att_out, O16);

  // ---- 5. y = O @ wo + wo_b + qn   (f32)
  wmma_gemm_kernel<true, true, false, true><<<gproj, 256, 0, stream>>>(
      O16, woT, wo_b, qnF, yF, nullptr, TOKENS, DIM, DIM);

  // ---- 6. LN2(y) -> hn f16
  ln2_kernel<<<TOKENS, 256, 0, stream>>>(yF, ln2_g, ln2_b, hn16);

  // ---- 7. h = gelu(hn @ w1 + b1)   (f16)
  dim3 gffn1(HIDDEN / 128, TOKENS / 128);
  wmma_gemm_kernel<true, false, true, false><<<gffn1, 256, 0, stream>>>(
      hn16, w1T, b1, nullptr, nullptr, h16, TOKENS, HIDDEN, DIM);

  // ---- 8. out = h @ w2 + b2 + y    (f32 -> d_out)
  wmma_gemm_kernel<true, true, false, true><<<gproj, 256, 0, stream>>>(
      h16, w2T, b2, yF, outF, nullptr, TOKENS, DIM, HIDDEN);
}
